// PointnetSA_MSG_15126874816948
// MI455X (gfx1250) — compile-verified, matrix operand order checked by reading, not run on hardware
//
#include <hip/hip_runtime.h>

typedef __attribute__((ext_vector_type(16))) _Float16 v16h;
typedef __attribute__((ext_vector_type(8)))  _Float16 v8h;
typedef __attribute__((ext_vector_type(8)))  float    v8f;

#define BATCH 8
#define NPTS  8192
#define NSAMP 1024   // S (npoint)
#define CFEAT 64

// ---------------------------------------------------------------------------
// 1) Farthest point sampling: one block (1024 threads) per batch. Each thread
//    keeps 8 points' coords + running min-dist in registers; argmax via wave
//    shuffle reduce + cross-wave smem reduce. 1024 sequential iterations.
// ---------------------------------------------------------------------------
__global__ __launch_bounds__(1024) void fps_kernel(const float* __restrict__ xyz,
                                                   int* __restrict__ fps_idx,
                                                   float* __restrict__ new_xyz,
                                                   float* __restrict__ d_out) {
    const int b = blockIdx.x;
    const int t = threadIdx.x;
    const float* base = xyz + (size_t)b * NPTS * 3;

    float px[8], py[8], pz[8], md[8];
#pragma unroll
    for (int i = 0; i < 8; ++i) {
        int j = t + (i << 10);
        px[i] = base[j * 3 + 0];
        py[i] = base[j * 3 + 1];
        pz[i] = base[j * 3 + 2];
        md[i] = 1e10f;
    }

    __shared__ float sv[32];
    __shared__ int   si[32];
    __shared__ int   s_cur;
    if (t == 0) s_cur = 0;
    __syncthreads();

    for (int it = 0; it < NSAMP; ++it) {
        int cur = s_cur;
        float cx = base[cur * 3 + 0];
        float cy = base[cur * 3 + 1];
        float cz = base[cur * 3 + 2];
        if (t == 0) {
            fps_idx[b * NSAMP + it] = cur;
            float* nw = new_xyz + (size_t)(b * NSAMP + it) * 3;
            nw[0] = cx; nw[1] = cy; nw[2] = cz;
            float* o = d_out + (size_t)(b * NSAMP + it) * 3;
            o[0] = cx; o[1] = cy; o[2] = cz;
        }
        float bv = -1.0f; int bi = 0;
#pragma unroll
        for (int i = 0; i < 8; ++i) {
            float dx = px[i] - cx, dy = py[i] - cy, dz = pz[i] - cz;
            float d = dx * dx + dy * dy + dz * dz;
            if (d < md[i]) md[i] = d;
            if (md[i] > bv) { bv = md[i]; bi = t + (i << 10); }
        }
        // wave32 reduce (prefer lower index on ties, matching jnp.argmax)
        for (int off = 16; off > 0; off >>= 1) {
            float ov = __shfl_down(bv, off, 32);
            int   oi = __shfl_down(bi, off, 32);
            if (ov > bv || (ov == bv && oi < bi)) { bv = ov; bi = oi; }
        }
        if ((t & 31) == 0) { sv[t >> 5] = bv; si[t >> 5] = bi; }
        __syncthreads();
        if (t == 0) {
            float gv = sv[0]; int gi = si[0];
            for (int w = 1; w < 32; ++w)
                if (sv[w] > gv || (sv[w] == gv && si[w] < gi)) { gv = sv[w]; gi = si[w]; }
            s_cur = gi;
        }
        __syncthreads();
    }
}

// ---------------------------------------------------------------------------
// 2) Ball query: one wave32 per (b,s) query; scan candidates in index order,
//    compact first `nsample` in-ball indices via ballot prefix sums.
// ---------------------------------------------------------------------------
__global__ void ball_query_kernel(const float* __restrict__ xyz,
                                  const float* __restrict__ new_xyz,
                                  int* __restrict__ idx, float r2, int nsample) {
    int wid  = (int)((blockIdx.x * blockDim.x + threadIdx.x) >> 5);
    int lane = threadIdx.x & 31;
    if (wid >= BATCH * NSAMP) return;
    int b = wid >> 10;
    const float* q = new_xyz + (size_t)wid * 3;
    float qx = q[0], qy = q[1], qz = q[2];
    const float* pts = xyz + (size_t)b * NPTS * 3;
    int* out = idx + (size_t)wid * nsample;

    int count = 0, firstIdx = 0; bool have = false;
    for (int basej = 0; basej < NPTS && count < nsample; basej += 32) {
        int j = basej + lane;
        float dx = pts[j * 3 + 0] - qx;
        float dy = pts[j * 3 + 1] - qy;
        float dz = pts[j * 3 + 2] - qz;
        bool within = (dx * dx + dy * dy + dz * dz) < r2;
        unsigned mask = (unsigned)__ballot(within);
        if (!have && mask) { firstIdx = basej + __builtin_ctz(mask); have = true; }
        int pre  = __popc(mask & ((1u << lane) - 1u));
        int slot = count + pre;
        if (within && slot < nsample) out[slot] = j;
        count += __popc(mask);
    }
    if (count > nsample) count = nsample;
    int padv = have ? firstIdx : 0;
    for (int sl = count + lane; sl < nsample; sl += 32) out[sl] = padv;
}

// ---------------------------------------------------------------------------
// 3) Group + f16 pack: row r=(b,s,k) of X[M x Kpad]: [rel_xyz(3) | feats(64) | 0pad]
// ---------------------------------------------------------------------------
__global__ void group_kernel(const float* __restrict__ xyz,
                             const float* __restrict__ feats,
                             const float* __restrict__ new_xyz,
                             const int* __restrict__ idx,
                             _Float16* __restrict__ X, int K, int Kpad, int Mtot) {
    int gid = blockIdx.x * blockDim.x + threadIdx.x;
    if (gid >= Mtot) return;
    int bs = gid / K;
    int b  = bs >> 10;
    int p  = idx[gid];
    const float* nw = new_xyz + (size_t)bs * 3;
    const float* pt = xyz + ((size_t)b * NPTS + p) * 3;
    _Float16* row = X + (size_t)gid * Kpad;
    row[0] = (_Float16)(pt[0] - nw[0]);
    row[1] = (_Float16)(pt[1] - nw[1]);
    row[2] = (_Float16)(pt[2] - nw[2]);
    const float* f = feats + (size_t)b * CFEAT * NPTS + p;
    for (int c = 0; c < CFEAT; ++c) row[3 + c] = (_Float16)f[(size_t)c * NPTS];
    for (int c = 3 + CFEAT; c < Kpad; ++c) row[c] = (_Float16)0.0f;
}

// Weight f32 (cout x cin) -> f16 (cout x Kpad), zero-padded K columns.
__global__ void convert_w_kernel(const float* __restrict__ w, _Float16* __restrict__ Wh,
                                 int cout, int cin, int Kpad) {
    int tid = blockIdx.x * blockDim.x + threadIdx.x;
    if (tid >= cout * Kpad) return;
    int n = tid / Kpad, k = tid % Kpad;
    Wh[tid] = (k < cin) ? (_Float16)w[n * cin + k] : (_Float16)0.0f;
}

__global__ void zero_f32_kernel(float* __restrict__ p, int n) {
    int tid = blockIdx.x * blockDim.x + threadIdx.x;
    if (tid < n) p[tid] = 0.0f;
}

// ---------------------------------------------------------------------------
// 4) WMMA GEMM v2: Y[M x Nc] = X[M x K] * Wh[Nc x K]^T + bias.
//    - Weights staged once per block into LDS (<= 32 KB), B fragments read
//      with ds_load_b128.
//    - One wave owns one 16-row M-tile and ALL NT column tiles: the A
//      fragment is loaded once per k-step (2x global_load_b128) and reused
//      across NT v_wmma_f32_16x16x32_f16 issues -> X is read exactly once.
//    - BatchNorm sum/sumsq reduced in LDS (ds_add_f32), one global atomic
//      per channel per block.
// ---------------------------------------------------------------------------
template <int NT>
__global__ __launch_bounds__(256) void gemm_wmma_f16(const _Float16* __restrict__ X,
                                                     const _Float16* __restrict__ Wh,
                                                     const float* __restrict__ bias,
                                                     _Float16* __restrict__ Y,
                                                     float* __restrict__ sum,
                                                     float* __restrict__ sumsq,
                                                     int M, int K) {
    constexpr int Nc = NT * 16;
    __shared__ _Float16 sW[128 * 128];   // max Cout x Kpad (32 KB)
    __shared__ float    sSum[128];
    __shared__ float    sSumSq[128];

    const int tid  = threadIdx.x;
    const int lane = tid & 31;
    const int wv   = tid >> 5;           // wave in block (0..7)
    const int MT   = M >> 4;
    const int mt   = blockIdx.x * 8 + wv;

    // cooperative weight stage: Nc*K halfs, 16B vector copies
    {
        const int total = Nc * K;        // multiple of 512
        for (int i = tid * 8; i < total; i += 256 * 8)
            *(v8h*)(&sW[i]) = *(const v8h*)(Wh + i);
        if (tid < 128) { sSum[tid] = 0.0f; sSumSq[tid] = 0.0f; }
    }
    __syncthreads();

    const int half_id = lane >> 4;
    const int l16     = lane & 15;

    if (mt < MT) {
        const _Float16* Arow = X + (size_t)(mt * 16 + l16) * K;

        v8f acc[NT];
#pragma unroll
        for (int nt = 0; nt < NT; ++nt) {
            float bval = bias[nt * 16 + l16];
#pragma unroll
            for (int r = 0; r < 8; ++r) acc[nt][r] = bval;
        }

        for (int kk = 0; kk < K; kk += 32) {
            if (kk + 32 < K)
                __builtin_prefetch(Arow + kk + 32, 0, 3);   // global_prefetch_b8
            v8h alo = *(const v8h*)(Arow + kk + 8 * half_id);
            v8h ahi = *(const v8h*)(Arow + kk + 16 + 8 * half_id);
            v16h a;
#pragma unroll
            for (int i = 0; i < 8; ++i) { a[i] = alo[i]; a[i + 8] = ahi[i]; }

#pragma unroll
            for (int nt = 0; nt < NT; ++nt) {
                const _Float16* Brow = &sW[(nt * 16 + l16) * K + kk];
                v8h blo = *(const v8h*)(Brow + 8 * half_id);       // ds_load_b128
                v8h bhi = *(const v8h*)(Brow + 16 + 8 * half_id);  // ds_load_b128
                v16h bb;
#pragma unroll
                for (int i = 0; i < 8; ++i) { bb[i] = blo[i]; bb[i + 8] = bhi[i]; }
                acc[nt] = __builtin_amdgcn_wmma_f32_16x16x32_f16(
                    false, a, false, bb, (short)0, acc[nt], false, false);
            }
        }

        // epilogue: store f16 + LDS-accumulate BN stats
#pragma unroll
        for (int nt = 0; nt < NT; ++nt) {
            float s1 = 0.f, s2 = 0.f;
#pragma unroll
            for (int r = 0; r < 8; ++r) { s1 += acc[nt][r]; s2 += acc[nt][r] * acc[nt][r]; }
            s1 += __shfl_xor(s1, 16, 32);
            s2 += __shfl_xor(s2, 16, 32);
            if (lane < 16) {
                atomicAdd(&sSum[nt * 16 + lane], s1);     // ds_add_f32
                atomicAdd(&sSumSq[nt * 16 + lane], s2);   // ds_add_f32
            }
            _Float16* Yp = Y + (size_t)(mt * 16 + 8 * half_id) * Nc + nt * 16 + l16;
#pragma unroll
            for (int r = 0; r < 8; ++r) Yp[(size_t)r * Nc] = (_Float16)acc[nt][r];
        }
    }

    __syncthreads();
    if (tid < Nc) {
        atomicAdd(&sum[tid], sSum[tid]);
        atomicAdd(&sumsq[tid], sSumSq[tid]);
    }
}

// Fold mean/var + gamma/beta into per-channel scale/shift.
__global__ void bn_finalize_kernel(const float* __restrict__ sum,
                                   const float* __restrict__ sumsq,
                                   const float* __restrict__ gamma,
                                   const float* __restrict__ beta,
                                   float* __restrict__ scale,
                                   float* __restrict__ shift, float invM) {
    int c = threadIdx.x;
    float mean = sum[c] * invM;
    float var  = sumsq[c] * invM - mean * mean;
    float sc   = gamma[c] * rsqrtf(var + 1e-5f);
    scale[c] = sc;
    shift[c] = beta[c] - mean * sc;
}

// In-place BN + ReLU on f16 GEMM output (feeds next GEMM).
__global__ void bn_apply_relu_kernel(_Float16* __restrict__ Y,
                                     const float* __restrict__ scale,
                                     const float* __restrict__ shift,
                                     long long total, int Cout) {
    long long i = (long long)blockIdx.x * blockDim.x + threadIdx.x;
    if (i >= total) return;
    int c = (int)(i % Cout);
    float v = (float)Y[i] * scale[c] + shift[c];
    Y[i] = (_Float16)fmaxf(v, 0.0f);
}

// BN + ReLU + max over K, writing to d_out (B, 320, S, 1) region.
__global__ void bn_max_out_kernel(const _Float16* __restrict__ Y,
                                  const float* __restrict__ scale,
                                  const float* __restrict__ shift,
                                  float* __restrict__ out,
                                  int K, int Cout, int c_off) {
    int tid = blockIdx.x * blockDim.x + threadIdx.x;
    if (tid >= BATCH * NSAMP * Cout) return;
    int n  = tid % Cout;
    int bs = tid / Cout;
    int b  = bs >> 10;
    int s  = bs & 1023;
    const _Float16* p = Y + (size_t)bs * K * Cout + n;
    float sc = scale[n], sh = shift[n];
    float m = 0.0f;  // ReLU outputs are >= 0
    for (int k = 0; k < K; ++k) {
        float v = (float)p[(size_t)k * Cout] * sc + sh;
        v = fmaxf(v, 0.0f);
        m = fmaxf(m, v);
    }
    out[(size_t)BATCH * NSAMP * 3 + ((size_t)b * 320 + c_off + n) * NSAMP + s] = m;
}

// ---------------------------------------------------------------------------
extern "C" void kernel_launch(void* const* d_in, const int* in_sizes, int n_in,
                              void* d_out, int out_size, void* d_ws, size_t ws_size,
                              hipStream_t stream) {
    const float* xyz   = (const float*)d_in[0];   // (8, 8192, 3)
    const float* feats = (const float*)d_in[1];   // (8, 64, 8192)
    float* out = (float*)d_out;

    char* ws = (char*)d_ws;
    // workspace layout (bytes)
    int*      fps_idx = (int*)(ws + 0);                      //  32 KB
    float*    nxyz    = (float*)(ws + 32768);                //  96 KB
    int*      idxbuf  = (int*)(ws + 131072);                 //   2 MB (max scale)
    float*    sumb    = (float*)(ws + 2228224);              // 128 f32
    float*    sumsqb  = (float*)(ws + 2228736);              // 128 f32
    float*    scaleb  = (float*)(ws + 2229248);              // 128 f32
    float*    shiftb  = (float*)(ws + 2229760);              // 128 f32
    _Float16* Wh      = (_Float16*)(ws + 2230272);           //  32 KB
    _Float16* BufX    = (_Float16*)(ws + 2263040);           // 134 MB (M3*128 f16)
    _Float16* BufY    = (_Float16*)(ws + 2263040 + 134217728ULL);

    // --- FPS + new_xyz (also first tuple output) ---
    fps_kernel<<<BATCH, 1024, 0, stream>>>(xyz, fps_idx, nxyz, out);

    const int   Ks[3]  = {16, 32, 64};
    const float Rs[3]  = {0.1f, 0.2f, 0.4f};
    const int   C1s[3] = {64, 64, 128};
    const int   C2s[3] = {64, 128, 128};
    const int   KPAD1  = 96;   // 67 padded to multiple of 32

    int c_off = 0;
    for (int sc = 0; sc < 3; ++sc) {
        const int K  = Ks[sc];
        const int M  = BATCH * NSAMP * K;
        const int C1 = C1s[sc];
        const int C2 = C2s[sc];
        const float* w1 = (const float*)d_in[2 + sc * 8 + 0];
        const float* b1 = (const float*)d_in[2 + sc * 8 + 1];
        const float* g1 = (const float*)d_in[2 + sc * 8 + 2];
        const float* e1 = (const float*)d_in[2 + sc * 8 + 3];
        const float* w2 = (const float*)d_in[2 + sc * 8 + 4];
        const float* b2 = (const float*)d_in[2 + sc * 8 + 5];
        const float* g2 = (const float*)d_in[2 + sc * 8 + 6];
        const float* e2 = (const float*)d_in[2 + sc * 8 + 7];

        // ball query: one wave per (b,s) => 8192 waves => 1024 blocks of 256
        ball_query_kernel<<<1024, 256, 0, stream>>>(xyz, nxyz, idxbuf,
                                                    Rs[sc] * Rs[sc], K);
        // group into X (M x 96) f16
        group_kernel<<<(M + 255) / 256, 256, 0, stream>>>(xyz, feats, nxyz, idxbuf,
                                                          BufX, K, KPAD1, M);
        // ---- layer 1: (67->C1) ----
        convert_w_kernel<<<(C1 * KPAD1 + 255) / 256, 256, 0, stream>>>(w1, Wh, C1, 67, KPAD1);
        zero_f32_kernel<<<1, 256, 0, stream>>>(sumb, 256);
        {
            int blocks = ((M >> 4) + 7) / 8;   // 8 M-tiles (waves) per block
            if (C1 == 64)
                gemm_wmma_f16<4><<<blocks, 256, 0, stream>>>(BufX, Wh, b1, BufY,
                                                             sumb, sumsqb, M, KPAD1);
            else
                gemm_wmma_f16<8><<<blocks, 256, 0, stream>>>(BufX, Wh, b1, BufY,
                                                             sumb, sumsqb, M, KPAD1);
        }
        bn_finalize_kernel<<<1, C1, 0, stream>>>(sumb, sumsqb, g1, e1,
                                                 scaleb, shiftb, 1.0f / (float)M);
        {
            long long total = (long long)M * C1;
            bn_apply_relu_kernel<<<(int)((total + 255) / 256), 256, 0, stream>>>(
                BufY, scaleb, shiftb, total, C1);
        }
        // ---- layer 2: (C1->C2), input BufY (M x C1), output BufX (M x C2) ----
        convert_w_kernel<<<(C2 * C1 + 255) / 256, 256, 0, stream>>>(w2, Wh, C2, C1, C1);
        zero_f32_kernel<<<1, 256, 0, stream>>>(sumb, 256);
        {
            int blocks = ((M >> 4) + 7) / 8;
            if (C2 == 64)
                gemm_wmma_f16<4><<<blocks, 256, 0, stream>>>(BufY, Wh, b2, BufX,
                                                             sumb, sumsqb, M, C1);
            else
                gemm_wmma_f16<8><<<blocks, 256, 0, stream>>>(BufY, Wh, b2, BufX,
                                                             sumb, sumsqb, M, C1);
        }
        bn_finalize_kernel<<<1, C2, 0, stream>>>(sumb, sumsqb, g2, e2,
                                                 scaleb, shiftb, 1.0f / (float)M);
        bn_max_out_kernel<<<(BATCH * NSAMP * C2 + 255) / 256, 256, 0, stream>>>(
            BufX, scaleb, shiftb, out, K, C2, c_off);
        c_off += C2;
    }
    (void)in_sizes; (void)n_in; (void)out_size; (void)ws_size;
}
